// LatentSDE_44908178047064
// MI455X (gfx1250) — matrix operand order repaired and checked by reading, not compiled, not run on hardware
//
#include <hip/hip_runtime.h>
#include <hip/hip_bf16.h>
#include <math.h>

// ---------------------------------------------------------------------------
// Latent SDE forward pass for MI455X (gfx1250, wave32, WMMA).
// B=256 split into 16 workgroups x 16 batches: both scans are batch-parallel,
// so each WGP runs its own independent recurrence with weights in LDS and the
// per-step matmuls done with v_wmma_f32_16x16x32_f16 (f16 in, f32 accumulate).
// Weights are pre-swizzled into WMMA fragment order so each per-step B-operand
// load is one contiguous 32B LDS read (ds_load_b128 x2).  Per-step global
// operands (xs / ctx / noise) are prefetched 2-3 steps ahead with
// global_prefetch_b8 to hide HBM latency on the sequential critical path.
// Workspace requirement: ~38 MB.
// ---------------------------------------------------------------------------

typedef __attribute__((ext_vector_type(16))) _Float16 v16h;
typedef __attribute__((ext_vector_type(8)))  _Float16 v8h;
typedef __attribute__((ext_vector_type(8)))  float    v8f;

#define T_LEN   1024
#define B_SZ    256
#define DT_F    0.01f
#define SQRT_DT_F 0.1f
#define LOG_2PI_F 1.837877066409345f

// ---- workspace byte offsets -----------------------------------------------
#define WS_WHH   0u            // 128x384 f16 : enc_Whh, WMMA-fragment order
#define WS_LINW  98304u        // 128x64  f16 : enc_linW, fragment order
#define WS_FW1C  114688u       // 64x128  f16 : fW1 ctx-part, fragment order
#define WS_FW2T  131072u       // 128x128 f16 : fW2, fragment order
#define WS_CTX   163840u       // T*B*64 f16  : encoder contexts
#define WS_ZS    33718272u     // T*B*4  f32  : latent path
#define WS_PATH  37912576u     // B f32       : per-batch sum of dlog
#define WS_ACC   37913600u     // f32[4]      : [0]=logqp0
#define WS_PART  37913856u     // 512*2 f32   : loss partials

// ---- helpers ---------------------------------------------------------------
__device__ __forceinline__ float sigmoidf_(float x) { return 1.0f / (1.0f + __expf(-x)); }
__device__ __forceinline__ float softplusf_(float x) { return (x > 20.f) ? x : log1pf(__expf(x)); }

__device__ __forceinline__ v8f wmma_f32_f16(v16h a, v16h b, v8f c) {
  return __builtin_amdgcn_wmma_f32_16x16x32_f16(false, a, false, b, (short)0, c, false, false);
}

// A fragment: 16x32 f16 tile from row-major [M][K] LDS buffer.
// Contiguous k order (k = 16*(lane/16) + s): one 16-half run per lane -> 2x ds_load_b128.
__device__ __forceinline__ v16h load_a16(const _Float16* A, int lda, int k0, int lane) {
  const _Float16* Ap = A + (lane & 15) * lda + k0 + ((lane < 16) ? 0 : 16);
  v8h lo = *(const v8h*)(Ap);
  v8h hi = *(const v8h*)(Ap + 8);
  return __builtin_shufflevector(lo, hi, 0, 1, 2, 3, 4, 5, 6, 7, 8, 9, 10, 11, 12, 13, 14, 15);
}

// B fragment from pre-swizzled buffer: fragment (nt*Ktiles+ks), lane-contiguous.
__device__ __forceinline__ v16h load_bfrag(const _Float16* P, int Ktiles, int ks, int nt, int lane) {
  return *(const v16h*)(P + ((size_t)(nt * Ktiles + ks) * 32 + lane) * 16);
}

// Packed-element -> (k, n) decode used by the prep kernel.
// i = ((nt*Ktiles+ks)*32 + lane)*16 + j ; k = ks*32 + (lane<16?0:16) + j ; n = nt*16 + (lane&15)
__device__ __forceinline__ void unpack_kn(int i, int KtilesMask, int KtilesShift, int* k, int* n) {
  int j = i & 15;
  int lane = (i >> 4) & 31;
  int f = i >> 9;
  int ks = f & KtilesMask;
  int nt = f >> KtilesShift;
  *k = ks * 32 + ((lane < 16) ? 0 : 16) + j;
  *n = nt * 16 + (lane & 15);
}

// ---------------------------------------------------------------------------
// K0: pre-swizzle weights into f16 WMMA-fragment order; zero accumulators.
// ---------------------------------------------------------------------------
__global__ void prep_kernel(const float* __restrict__ Whh, const float* __restrict__ linW,
                            const float* __restrict__ fW1, const float* __restrict__ fW2,
                            _Float16* __restrict__ wWHH, _Float16* __restrict__ wLINW,
                            _Float16* __restrict__ wFW1C, _Float16* __restrict__ wFW2T,
                            float* __restrict__ acc) {
  int gid = blockIdx.x * blockDim.x + threadIdx.x;
  int nth = gridDim.x * blockDim.x;
  int k, n;
  // enc_Whh (384,128): B[k][n] = Whh[n][k], K=128 (Ktiles=4), N=384
  for (int i = gid; i < 384 * 128; i += nth) { unpack_kn(i, 3, 2, &k, &n); wWHH[i] = (_Float16)Whh[n * 128 + k]; }
  // enc_linW (64,128): B[k][n] = linW[n][k], K=128 (Ktiles=4), N=64
  for (int i = gid; i < 64 * 128; i += nth)  { unpack_kn(i, 3, 2, &k, &n); wLINW[i] = (_Float16)linW[n * 128 + k]; }
  // fW1 ctx part (128,68): B[k][n] = fW1[n][4+k], K=64 (Ktiles=2), N=128
  for (int i = gid; i < 128 * 64; i += nth)  { unpack_kn(i, 1, 1, &k, &n); wFW1C[i] = (_Float16)fW1[n * 68 + 4 + k]; }
  // fW2 (128,128): B[k][n] = fW2[n][k], K=128 (Ktiles=4), N=128
  for (int i = gid; i < 128 * 128; i += nth) { unpack_kn(i, 3, 2, &k, &n); wFW2T[i] = (_Float16)fW2[n * 128 + k]; }
  if (gid < 4) acc[gid] = 0.f;
}

// ---------------------------------------------------------------------------
// K1: GRU encoder scan (reverse time).  grid = 16 blocks x 256 threads.
// Block g owns batches [16g, 16g+16).  Emits ctx[t][b][0..63] as f16.
// LDS layout (bytes):
#define G_WHH  0u        // 98304  f16 fragment-order 128x384
#define G_LINW 98304u    // 16384  f16 fragment-order 128x64
#define G_WIH  114688u   // 7680   f32 384x5
#define G_BIH  122368u   // 1536
#define G_BHH  123904u   // 1536
#define G_LINB 125440u   // 256
#define G_H    125696u   // 8192   f32 16x128 hidden state
#define G_HH   133888u   // 4096   f16 16x128 hidden state (WMMA A)
#define G_GI   137984u   // 24576  f32 16x384 input gates
#define G_GH   162560u   // 24576  f32 16x384 hidden gates
#define G_XS   187136u   // 256    f32 16x4
#define G_LDS_BYTES 187392u
// ---------------------------------------------------------------------------
__global__ void gru_scan_kernel(const float* __restrict__ xs, const float* __restrict__ Wih,
                                const float* __restrict__ bih, const float* __restrict__ bhh,
                                const float* __restrict__ linb, const float* __restrict__ cW,
                                const float* __restrict__ cb,
                                const _Float16* __restrict__ wWHH, const _Float16* __restrict__ wLINW,
                                _Float16* __restrict__ ctx_out) {
  extern __shared__ char smem[];
  _Float16* sWHH  = (_Float16*)(smem + G_WHH);
  _Float16* sLINW = (_Float16*)(smem + G_LINW);
  float* sWIH  = (float*)(smem + G_WIH);
  float* sBIH  = (float*)(smem + G_BIH);
  float* sBHH  = (float*)(smem + G_BHH);
  float* sLINB = (float*)(smem + G_LINB);
  float* sH    = (float*)(smem + G_H);
  _Float16* sHH = (_Float16*)(smem + G_HH);
  float* sGI   = (float*)(smem + G_GI);
  float* sGH   = (float*)(smem + G_GH);
  float* sXS   = (float*)(smem + G_XS);

  const int tid = threadIdx.x;
  const int lane = tid & 31;
  const int wave = tid >> 5;
  const int b0 = blockIdx.x * 16;

  // vectorized weight preload (all buffers 16B aligned)
  for (int i = tid; i < 6144; i += 256) ((uint4*)sWHH)[i] = ((const uint4*)wWHH)[i];
  for (int i = tid; i < 1024; i += 256) ((uint4*)sLINW)[i] = ((const uint4*)wLINW)[i];
  for (int i = tid; i < 480; i += 256)  ((uint4*)sWIH)[i] = ((const uint4*)Wih)[i];
  for (int i = tid; i < 96; i += 256)   { ((uint4*)sBIH)[i] = ((const uint4*)bih)[i];
                                          ((uint4*)sBHH)[i] = ((const uint4*)bhh)[i]; }
  for (int i = tid; i < 16; i += 256)   ((uint4*)sLINB)[i] = ((const uint4*)linb)[i];
  uint4 z4; z4.x = 0u; z4.y = 0u; z4.z = 0u; z4.w = 0u;
  for (int i = tid; i < 512; i += 256)  ((uint4*)sH)[i] = z4;
  for (int i = tid; i < 256; i += 256)  ((uint4*)sHH)[i] = z4;
  const float cw0 = cW[0], cw1 = cW[1], cw2 = cW[2], cw3 = cW[3], cb0 = cb[0];
  __syncthreads();

  for (int t = T_LEN - 1; t >= 0; --t) {
    // P1: stage this step's observations; prefetch observations 2 steps ahead
    if (tid < 16) ((uint4*)sXS)[tid] = ((const uint4*)(xs + ((size_t)t * B_SZ + b0) * 4))[tid];
    if (t >= 2 && tid < 2)
      __builtin_prefetch(xs + ((size_t)(t - 2) * B_SZ + b0) * 4 + tid * 32, 0, 1);
    __syncthreads();
    // P2: gi = xus @ Wih.T + bih  (K=5, VALU)
    for (int i = tid; i < 16 * 384; i += 256) {
      int b = i / 384, o = i % 384;
      const float* xb = sXS + b * 4;
      float us = xb[0] * cw0 + xb[1] * cw1 + xb[2] * cw2 + xb[3] * cw3 + cb0;
      const float* wr = sWIH + o * 5;
      sGI[i] = sBIH[o] + xb[0] * wr[0] + xb[1] * wr[1] + xb[2] * wr[2] + xb[3] * wr[3] + us * wr[4];
    }
    __syncthreads();
    // P3: gh = h @ Whh.T via WMMA (M=16, N=384, K=128); 3 tiles per wave, unrolled
    {
      v16h aF[4];
#pragma unroll
      for (int s = 0; s < 4; ++s) aF[s] = load_a16(sHH, 128, 32 * s, lane);
#pragma unroll
      for (int u = 0; u < 3; ++u) {
        const int nt = wave + 8 * u;
        v8f acc = {};
#pragma unroll
        for (int s = 0; s < 4; ++s)
          acc = wmma_f32_f16(aF[s], load_bfrag(sWHH, 4, s, nt, lane), acc);
        int col = nt * 16 + (lane & 15);
        int rbase = (lane < 16) ? 0 : 8;
#pragma unroll
        for (int r2 = 0; r2 < 8; ++r2) sGH[(size_t)(rbase + r2) * 384 + col] = acc[r2];
      }
    }
    __syncthreads();
    // P4: gates + hidden update
    for (int i = tid; i < 16 * 128; i += 256) {
      int b = i >> 7, kk = i & 127;
      float gi_r = sGI[b * 384 + kk];
      float gh_r = sGH[b * 384 + kk] + sBHH[kk];
      float gi_z = sGI[b * 384 + 128 + kk];
      float gh_z = sGH[b * 384 + 128 + kk] + sBHH[128 + kk];
      float gi_n = sGI[b * 384 + 256 + kk];
      float gh_n = sGH[b * 384 + 256 + kk] + sBHH[256 + kk];
      float rg = sigmoidf_(gi_r + gh_r);
      float zg = sigmoidf_(gi_z + gh_z);
      float ng = tanhf(gi_n + rg * gh_n);
      float hn = (1.f - zg) * ng + zg * sH[i];
      sH[i] = hn;
      sHH[i] = (_Float16)hn;
    }
    __syncthreads();
    // P5: ctx[t] = h @ linW.T + linb via WMMA (M=16, N=64, K=128), f16 out
    if (wave < 4) {
      v8f acc = {};
#pragma unroll
      for (int s = 0; s < 4; ++s)
        acc = wmma_f32_f16(load_a16(sHH, 128, 32 * s, lane),
                           load_bfrag(sLINW, 4, s, wave, lane), acc);
      int col = wave * 16 + (lane & 15);
      int rbase = (lane < 16) ? 0 : 8;
#pragma unroll
      for (int r2 = 0; r2 < 8; ++r2) {
        int b = rbase + r2;
        ctx_out[((size_t)t * B_SZ + b0 + b) * 64 + col] = (_Float16)(acc[r2] + sLINB[col]);
      }
    }
    __syncthreads();
  }
}

// ---------------------------------------------------------------------------
// K2: q(z0), sample z0, KL term.  1 block x 256 threads (one per batch).
// ---------------------------------------------------------------------------
__global__ void z0_kernel(const _Float16* __restrict__ ctx0, const float* __restrict__ qzW,
                          const float* __restrict__ qzb, const float* __restrict__ z0_eps,
                          const float* __restrict__ pz0_mean, const float* __restrict__ pz0_logstd,
                          float* __restrict__ zs, float* __restrict__ acc) {
  __shared__ float red[256];
  const int b = threadIdx.x;
  float qm[4], ql[4];
#pragma unroll
  for (int l = 0; l < 4; ++l) {
    float sm = qzb[l], sl = qzb[4 + l];
    for (int n = 0; n < 64; ++n) {
      float c = (float)ctx0[b * 64 + n];
      sm += c * qzW[l * 64 + n];
      sl += c * qzW[(4 + l) * 64 + n];
    }
    qm[l] = sm; ql[l] = sl;
  }
  float kls = 0.f;
#pragma unroll
  for (int l = 0; l < 4; ++l) {
    float z0v = qm[l] + 0.2f * __expf(ql[l]) * z0_eps[b * 4 + l];
    zs[b * 4 + l] = z0v;  // zs[t=0]
    float s1 = __expf(ql[l]), s2 = __expf(pz0_logstd[l]);
    float dm = qm[l] - pz0_mean[l];
    kls += pz0_logstd[l] - ql[l] + (s1 * s1 + dm * dm) / (2.f * s2 * s2) - 0.5f;
  }
  red[b] = kls; __syncthreads();
  for (int s = 128; s > 0; s >>= 1) { if (b < s) red[b] += red[b + s]; __syncthreads(); }
  if (b == 0) acc[0] = red[0] / (float)B_SZ;
}

// ---------------------------------------------------------------------------
// K3: SDE scan. grid = 16 blocks x 256 threads, block g owns 16 batches.
// LDS layout (bytes):
#define S_W1   0u       // 16384  f16 fragment-order 64x128
#define S_W2   16384u   // 32768  f16 fragment-order 128x128
#define S_FW3  49152u   // 2048   f32 4x128
#define S_W1Z  51200u   // 2048   f32 128x4    fW1 z-part [o][j]
#define S_FB1  53248u   // 512
#define S_FB2  53760u   // 512
#define S_GW1  54272u   // 2048
#define S_GB1  56320u   // 2048
#define S_GW2  58368u   // 2048
#define S_C    60416u   // 2048   f16 16x64
#define S_H1H  62464u   // 4096   f16 16x128
#define S_H2   66560u   // 8192   f32 16x128
#define S_Z    74752u   // 256
#define S_ZN   75008u   // 256
#define S_FZ   75264u   // 256
#define S_HZ   75520u   // 256
#define S_GZ   75776u   // 256
#define S_UR   76032u   // 256
#define S_U    76288u   // 64
#define S_DLOG 76352u   // 64
#define S_CST  76416u   // 256
#define S_LDS_BYTES 76672u
// const-pool float offsets
#define CST_PW  0
#define CST_PB  16
#define CST_CW  20
#define CST_CB  24
#define CST_HW  25
#define CST_HB  45
#define CST_GB2 49
#define CST_FB3 53
// ---------------------------------------------------------------------------
__global__ void sde_scan_kernel(const _Float16* __restrict__ ctx, const _Float16* __restrict__ wFW1C,
                                const _Float16* __restrict__ wFW2T,
                                const float* __restrict__ fW1, const float* __restrict__ fb1,
                                const float* __restrict__ fb2, const float* __restrict__ fW3,
                                const float* __restrict__ fb3, const float* __restrict__ hW,
                                const float* __restrict__ hb, const float* __restrict__ cW,
                                const float* __restrict__ cb, const float* __restrict__ pW,
                                const float* __restrict__ pb, const float* __restrict__ gW1,
                                const float* __restrict__ gb1, const float* __restrict__ gW2,
                                const float* __restrict__ gb2, const float* __restrict__ bm,
                                float* __restrict__ zs, float* __restrict__ pathsum) {
  extern __shared__ char smem[];
  _Float16* sW1 = (_Float16*)(smem + S_W1);
  _Float16* sW2 = (_Float16*)(smem + S_W2);
  float* sFW3 = (float*)(smem + S_FW3);
  float* sW1Z = (float*)(smem + S_W1Z);
  float* sFB1 = (float*)(smem + S_FB1);
  float* sFB2 = (float*)(smem + S_FB2);
  float* sGW1 = (float*)(smem + S_GW1);
  float* sGB1 = (float*)(smem + S_GB1);
  float* sGW2 = (float*)(smem + S_GW2);
  _Float16* sC = (_Float16*)(smem + S_C);
  _Float16* sH1H = (_Float16*)(smem + S_H1H);
  float* sH2 = (float*)(smem + S_H2);
  float* sZ = (float*)(smem + S_Z);
  float* sZn = (float*)(smem + S_ZN);
  float* sFz = (float*)(smem + S_FZ);
  float* sHz = (float*)(smem + S_HZ);
  float* sGz = (float*)(smem + S_GZ);
  float* sUr = (float*)(smem + S_UR);
  float* sU = (float*)(smem + S_U);
  float* sDlog = (float*)(smem + S_DLOG);
  float* sCst = (float*)(smem + S_CST);

  const int tid = threadIdx.x;
  const int lane = tid & 31;
  const int wave = tid >> 5;
  const int b0 = blockIdx.x * 16;

  for (int i = tid; i < 1024; i += 256) ((uint4*)sW1)[i] = ((const uint4*)wFW1C)[i];
  for (int i = tid; i < 2048; i += 256) ((uint4*)sW2)[i] = ((const uint4*)wFW2T)[i];
  for (int i = tid; i < 128; i += 256) {
    ((uint4*)sFW3)[i] = ((const uint4*)fW3)[i];
    ((uint4*)sGW1)[i] = ((const uint4*)gW1)[i];
    ((uint4*)sGB1)[i] = ((const uint4*)gb1)[i];
    ((uint4*)sGW2)[i] = ((const uint4*)gW2)[i];
  }
  for (int i = tid; i < 32; i += 256) {
    ((uint4*)sFB1)[i] = ((const uint4*)fb1)[i];
    ((uint4*)sFB2)[i] = ((const uint4*)fb2)[i];
  }
  for (int i = tid; i < 512; i += 256) { int o = i >> 2, j = i & 3; sW1Z[i] = fW1[o * 68 + j]; }
  if (tid < 16) sCst[CST_PW + tid] = pW[tid];
  if (tid < 20) sCst[CST_HW + tid] = hW[tid];
  if (tid < 4) {
    sCst[CST_PB + tid] = pb[tid]; sCst[CST_CW + tid] = cW[tid];
    sCst[CST_HB + tid] = hb[tid]; sCst[CST_GB2 + tid] = gb2[tid];
    sCst[CST_FB3 + tid] = fb3[tid];
  }
  if (tid == 0) sCst[CST_CB] = cb[0];
  if (tid < 64) sZ[tid] = zs[(size_t)(b0 + (tid >> 2)) * 4 + (tid & 3)];  // z0
  if (tid < 16) sDlog[tid] = 0.f;
  __syncthreads();

  for (int k = 0; k < T_LEN - 1; ++k) {
    // A: stage ctx[k+1] (contiguous 2KB slice) ; u_ctl from current z.
    // Prefetch ctx 3 steps and noise 2 steps ahead to hide HBM latency.
    if (tid < 128)
      ((uint4*)sC)[tid] = ((const uint4*)(ctx + ((size_t)(k + 1) * B_SZ + b0) * 64))[tid];
    if (k + 3 < T_LEN && tid >= 128 && tid < 144)
      __builtin_prefetch(ctx + ((size_t)(k + 3) * B_SZ + b0) * 64 + (tid - 128) * 64, 0, 1);
    if (k + 2 < T_LEN - 1 && tid >= 144 && tid < 146)
      __builtin_prefetch(bm + ((size_t)(k + 2) * B_SZ + b0) * 4 + (tid - 144) * 32, 0, 1);
    if (tid < 16) {
      float u = sCst[CST_CB];
#pragma unroll
      for (int d = 0; d < 4; ++d) {
        float pd = sCst[CST_PB + d];
#pragma unroll
        for (int l = 0; l < 4; ++l) pd += sZ[tid * 4 + l] * sCst[CST_PW + d * 4 + l];
        u += sCst[CST_CW + d] * pd;
      }
      sU[tid] = u;
    }
    __syncthreads();
    // B: h1 = softplus([z,c] @ fW1.T + fb1)   (ctx part via WMMA, z part fused)
    {
      v8f acc = {};
      acc = wmma_f32_f16(load_a16(sC, 64, 0, lane),  load_bfrag(sW1, 2, 0, wave, lane), acc);
      acc = wmma_f32_f16(load_a16(sC, 64, 32, lane), load_bfrag(sW1, 2, 1, wave, lane), acc);
      int col = wave * 16 + (lane & 15);
      int rbase = (lane < 16) ? 0 : 8;
#pragma unroll
      for (int r2 = 0; r2 < 8; ++r2) {
        int b = rbase + r2;
        float v = acc[r2] + sFB1[col];
#pragma unroll
        for (int j = 0; j < 4; ++j) v += sZ[b * 4 + j] * sW1Z[col * 4 + j];
        sH1H[b * 128 + col] = (_Float16)softplusf_(v);
      }
    }
    __syncthreads();
    // D: h2 = softplus(h1 @ fW2.T + fb2) via WMMA (K=128)
    {
      v8f acc = {};
#pragma unroll
      for (int s = 0; s < 4; ++s)
        acc = wmma_f32_f16(load_a16(sH1H, 128, 32 * s, lane),
                           load_bfrag(sW2, 4, s, wave, lane), acc);
      int col = wave * 16 + (lane & 15);
      int rbase = (lane < 16) ? 0 : 8;
#pragma unroll
      for (int r2 = 0; r2 < 8; ++r2)
        sH2[(rbase + r2) * 128 + col] = softplusf_(acc[r2] + sFB2[col]);
    }
    __syncthreads();
    // F: fz (tid<64), gz (64..127), hz (128..191)
    if (tid < 64) {
      int b = tid >> 2, l = tid & 3;
      float s = sCst[CST_FB3 + l];
      for (int k2 = 0; k2 < 128; ++k2) s += sH2[b * 128 + k2] * sFW3[l * 128 + k2];
      sFz[tid] = s;
    } else if (tid < 128) {
      int i2 = tid - 64; int b = i2 >> 2, l = i2 & 3;
      float zv = sZ[b * 4 + l];
      float g = sCst[CST_GB2 + l];
      for (int k2 = 0; k2 < 128; ++k2)
        g += softplusf_(zv * sGW1[l * 128 + k2] + sGB1[l * 128 + k2]) * sGW2[l * 128 + k2];
      sGz[i2] = sigmoidf_(g);
    } else if (tid < 192) {
      int i2 = tid - 128; int b = i2 >> 2, l = i2 & 3;
      float h = sCst[CST_HB + l];
#pragma unroll
      for (int j = 0; j < 4; ++j) h += sZ[b * 4 + j] * sCst[CST_HW + l * 5 + j];
      h += sU[b] * sCst[CST_HW + l * 5 + 4];
      sHz[i2] = h;
    }
    __syncthreads();
    // G: girsanov residual + Euler-Maruyama step
    if (tid < 64) {
      int b = tid >> 2, l = tid & 3;
      float fzv = sFz[tid], hzv = sHz[tid], gzv = sGz[tid];
      float ur = (fzv - hzv) / gzv;
      sUr[tid] = ur * ur;
      float nz = bm[((size_t)k * B_SZ + b0 + b) * 4 + l];
      float zn = sZ[tid] + fzv * DT_F + gzv * SQRT_DT_F * nz;
      zs[((size_t)(k + 1) * B_SZ + b0 + b) * 4 + l] = zn;
      sZn[tid] = zn;
    }
    __syncthreads();
    // H: accumulate dlog, commit z
    if (tid < 16)
      sDlog[tid] += 0.5f * DT_F * (sUr[tid * 4] + sUr[tid * 4 + 1] + sUr[tid * 4 + 2] + sUr[tid * 4 + 3]);
    if (tid < 64) sZ[tid] = sZn[tid];
    __syncthreads();
  }
  if (tid < 16) pathsum[b0 + tid] = sDlog[tid];
}

// ---------------------------------------------------------------------------
// K4: decoder + loss partials over T*B items.  512 blocks x 256 threads.
// ---------------------------------------------------------------------------
__global__ void loss_kernel(const float* __restrict__ xs, const float* __restrict__ zs,
                            const float* __restrict__ pW, const float* __restrict__ pb,
                            const float* __restrict__ noise_std, float* __restrict__ part) {
  __shared__ float r1[256];
  __shared__ float r2[256];
  const int tid = threadIdx.x;
  size_t gid = (size_t)blockIdx.x * 256 + tid;
  const float scale = noise_std[0];
  const float inv = 1.f / scale;
  const float ls = __logf(scale);
  float pwv[16], pbv[4];
#pragma unroll
  for (int i = 0; i < 16; ++i) pwv[i] = pW[i];
#pragma unroll
  for (int i = 0; i < 4; ++i) pbv[i] = pb[i];
  float nrm_s = 0.f, lp_s = 0.f;
  const size_t total = (size_t)T_LEN * B_SZ;
  for (size_t i = gid; i < total; i += (size_t)512 * 256) {
    float z0v = zs[i * 4 + 0], z1v = zs[i * 4 + 1], z2v = zs[i * 4 + 2], z3v = zs[i * 4 + 3];
    float sn = 0.f, sq = 0.f;
#pragma unroll
    for (int d = 0; d < 4; ++d) {
      float xv = pbv[d] + pwv[d * 4 + 0] * z0v + pwv[d * 4 + 1] * z1v + pwv[d * 4 + 2] * z2v + pwv[d * 4 + 3] * z3v;
      sn += xv * xv;
      float df = (xs[i * 4 + d] - xv) * inv;
      sq += df * df;
    }
    nrm_s += sqrtf(sn);
    lp_s += -0.5f * sq - 4.f * ls - 2.f * LOG_2PI_F;
  }
  r1[tid] = nrm_s; r2[tid] = lp_s; __syncthreads();
  for (int s = 128; s > 0; s >>= 1) {
    if (tid < s) { r1[tid] += r1[tid + s]; r2[tid] += r2[tid + s]; }
    __syncthreads();
  }
  if (tid == 0) { part[2 * blockIdx.x] = r1[0]; part[2 * blockIdx.x + 1] = r2[0]; }
}

// ---------------------------------------------------------------------------
// K5: finalize the three scalars (deterministic reduction, no atomics).
// ---------------------------------------------------------------------------
__global__ void finalize_kernel(const float* __restrict__ part, const float* __restrict__ pathsum,
                                const float* __restrict__ acc, float* __restrict__ out) {
  __shared__ float r1[256], r2[256], r3[256];
  const int tid = threadIdx.x;
  float pn = 0.f, pl = 0.f;
  for (int i = tid; i < 512; i += 256) { pn += part[2 * i]; pl += part[2 * i + 1]; }
  r1[tid] = pn; r2[tid] = pl; r3[tid] = pathsum[tid];
  __syncthreads();
  for (int s = 128; s > 0; s >>= 1) {
    if (tid < s) { r1[tid] += r1[tid + s]; r2[tid] += r2[tid + s]; r3[tid] += r3[tid + s]; }
    __syncthreads();
  }
  if (tid == 0) {
    out[0] = r2[0] / (float)B_SZ;                        // log_pxs
    out[1] = acc[0] + 35.f * (r3[0] / (float)B_SZ);      // logqp0 + 35*logqp_path
    out[2] = r1[0] / (float)B_SZ;                        // r
  }
}

// ---------------------------------------------------------------------------
extern "C" void kernel_launch(void* const* d_in, const int* in_sizes, int n_in,
                              void* d_out, int out_size, void* d_ws, size_t ws_size,
                              hipStream_t stream) {
  (void)in_sizes; (void)n_in; (void)out_size; (void)ws_size;
  const float* xs        = (const float*)d_in[0];
  const float* noise_std = (const float*)d_in[2];
  const float* z0_eps    = (const float*)d_in[3];
  const float* bm_noise  = (const float*)d_in[4];
  const float* enc_Wih   = (const float*)d_in[5];
  const float* enc_bih   = (const float*)d_in[6];
  const float* enc_Whh   = (const float*)d_in[7];
  const float* enc_bhh   = (const float*)d_in[8];
  const float* enc_linW  = (const float*)d_in[9];
  const float* enc_linb  = (const float*)d_in[10];
  const float* qzW       = (const float*)d_in[11];
  const float* qzb       = (const float*)d_in[12];
  const float* fW1       = (const float*)d_in[13];
  const float* fb1       = (const float*)d_in[14];
  const float* fW2       = (const float*)d_in[15];
  const float* fb2       = (const float*)d_in[16];
  const float* fW3       = (const float*)d_in[17];
  const float* fb3       = (const float*)d_in[18];
  const float* cW        = (const float*)d_in[19];
  const float* cb        = (const float*)d_in[20];
  const float* hW        = (const float*)d_in[21];
  const float* hb        = (const float*)d_in[22];
  const float* gW1       = (const float*)d_in[23];
  const float* gb1       = (const float*)d_in[24];
  const float* gW2       = (const float*)d_in[25];
  const float* gb2       = (const float*)d_in[26];
  const float* pW        = (const float*)d_in[27];
  const float* pb        = (const float*)d_in[28];
  const float* pz0_mean  = (const float*)d_in[29];
  const float* pz0_lstd  = (const float*)d_in[30];

  char* ws = (char*)d_ws;
  _Float16* wWHH  = (_Float16*)(ws + WS_WHH);
  _Float16* wLINW = (_Float16*)(ws + WS_LINW);
  _Float16* wFW1C = (_Float16*)(ws + WS_FW1C);
  _Float16* wFW2T = (_Float16*)(ws + WS_FW2T);
  _Float16* ctxw  = (_Float16*)(ws + WS_CTX);
  float* zsw   = (float*)(ws + WS_ZS);
  float* pathw = (float*)(ws + WS_PATH);
  float* accw  = (float*)(ws + WS_ACC);
  float* partw = (float*)(ws + WS_PART);

  (void)hipFuncSetAttribute(reinterpret_cast<const void*>(gru_scan_kernel),
                            hipFuncAttributeMaxDynamicSharedMemorySize, (int)G_LDS_BYTES);
  (void)hipFuncSetAttribute(reinterpret_cast<const void*>(sde_scan_kernel),
                            hipFuncAttributeMaxDynamicSharedMemorySize, (int)S_LDS_BYTES);

  prep_kernel<<<32, 256, 0, stream>>>(enc_Whh, enc_linW, fW1, fW2, wWHH, wLINW, wFW1C, wFW2T, accw);
  gru_scan_kernel<<<16, 256, G_LDS_BYTES, stream>>>(xs, enc_Wih, enc_bih, enc_bhh, enc_linb,
                                                    cW, cb, wWHH, wLINW, ctxw);
  z0_kernel<<<1, 256, 0, stream>>>(ctxw, qzW, qzb, z0_eps, pz0_mean, pz0_lstd, zsw, accw);
  sde_scan_kernel<<<16, 256, S_LDS_BYTES, stream>>>(ctxw, wFW1C, wFW2T, fW1, fb1, fb2, fW3, fb3,
                                                    hW, hb, cW, cb, pW, pb, gW1, gb1, gW2, gb2,
                                                    bm_noise, zsw, pathw);
  loss_kernel<<<512, 256, 0, stream>>>(xs, zsw, pW, pb, noise_std, partw);
  finalize_kernel<<<1, 256, 0, stream>>>(partw, pathw, accw, (float*)d_out);
}
